// TRANSFORMER_GATNet_33629593928258
// MI455X (gfx1250) — compile-verified, matrix operand order checked by reading, not run on hardware
//
#include <hip/hip_runtime.h>
#include <hip/hip_bf16.h>
#include <math.h>

#define Bc    32
#define MAXNc 512
#define Dc    256
#define Hc    8
#define DHc   32
#define FFc   1024
#define TQc   (Bc*MAXNc)

typedef __attribute__((ext_vector_type(16))) _Float16 v16h;
typedef __attribute__((ext_vector_type(8)))  _Float16 v8h;
typedef __attribute__((ext_vector_type(8)))  float    v8f;

union V16U { v16h v; v8h h[2]; };

// ======================= small elementwise kernels =======================
__global__ void fill_f_kernel(float* p, float v, long long n){
  long long i = (long long)blockIdx.x*blockDim.x + threadIdx.x;
  if (i < n) p[i] = v;
}
__global__ void fill_i_kernel(int* p, int v, long long n){
  long long i = (long long)blockIdx.x*blockDim.x + threadIdx.x;
  if (i < n) p[i] = v;
}
__global__ void bias_act_kernel(float* y, const float* __restrict__ bias,
                                long long n, int F, int act){
  long long i = (long long)blockIdx.x*blockDim.x + threadIdx.x;
  if (i >= n) return;
  float v = y[i] + bias[(int)(i % F)];
  if (act == 1)      v = fmaxf(v, 0.0f);
  else if (act == 2) v = (v > 0.0f) ? v : (__expf(v) - 1.0f);  // ELU(1)
  y[i] = v;
}

// ======================= generic batched WMMA GEMM =======================
// C[z] = act( alpha * A[z] (MxK) * B[z] (KxN, or NxK if BT) + bias )
// Block tile 128x64, BK=32, 8 waves (4x2), wave tile 32x32 -> 4 WMMAs/wave/k-step.
// Double-buffered LDS, single barrier per k-step; next-tile global fetch overlaps WMMA.
// LDS is K-contiguous for BOTH operands: every 8-half fragment group is one
// aligned ds_load_b128 (row stride 40 halves = 80B, 16B-aligned).
template<int ACT>
__device__ inline float act_apply(float x){
  if (ACT == 1) return fmaxf(x, 0.0f);
  return x;
}

__device__ __forceinline__ v8h cvt_pack8(float4 a, float4 b){
  v8h h;
  h[0]=(_Float16)a.x; h[1]=(_Float16)a.y; h[2]=(_Float16)a.z; h[3]=(_Float16)a.w;
  h[4]=(_Float16)b.x; h[5]=(_Float16)b.y; h[6]=(_Float16)b.z; h[7]=(_Float16)b.w;
  return h;
}
__device__ __forceinline__ v8h zero8(){
  v8h h;
#pragma unroll
  for (int j = 0; j < 8; ++j) h[j] = (_Float16)0.0f;
  return h;
}

// fetch 16 k's of one A row into two v8h (with prefetch of next k-tile)
__device__ __forceinline__ void fetchA(const float* __restrict__ A, int lda, int M,
                                       int m0, int ar, int ak, int k0, int K,
                                       v8h& h0, v8h& h1){
  int gm = m0 + ar;
  if (gm < M) {
    const float* gp = A + (long long)gm*lda + k0 + ak;
    if (k0 + 32 < K) __builtin_prefetch(gp + 32, 0, 3);
    float4 f0 = ((const float4*)gp)[0];
    float4 f1 = ((const float4*)gp)[1];
    float4 f2 = ((const float4*)gp)[2];
    float4 f3 = ((const float4*)gp)[3];
    h0 = cvt_pack8(f0, f1);
    h1 = cvt_pack8(f2, f3);
  } else { h0 = zero8(); h1 = zero8(); }
}

// fetch 8 elements of B into one v8h.
// BT:  8 k's of row n=btn          (contiguous, stored contiguous)
// !BT: 8 n's of k-row bnr          (contiguous load, scattered store)
template<bool BT>
__device__ __forceinline__ v8h fetchB(const float* __restrict__ Bm, int ldb, int N,
                                      int n0, int btn, int btk, int bnr, int bnn,
                                      int k0, int K){
  if (BT) {
    int gn = n0 + btn;
    if (gn < N) {
      const float* gp = Bm + (long long)gn*ldb + k0 + btk;
      if (k0 + 32 < K) __builtin_prefetch(gp + 32, 0, 3);
      float4 f0 = ((const float4*)gp)[0];
      float4 f1 = ((const float4*)gp)[1];
      return cvt_pack8(f0, f1);
    }
    return zero8();
  } else {
    if (n0 + bnn + 8 <= N) {
      const float* gp = Bm + (long long)(k0 + bnr)*ldb + n0 + bnn;
      float4 f0 = ((const float4*)gp)[0];
      float4 f1 = ((const float4*)gp)[1];
      return cvt_pack8(f0, f1);
    }
    v8h h;
#pragma unroll
    for (int j = 0; j < 8; ++j) {
      int gn = n0 + bnn + j;
      h[j] = (_Float16)((gn < N) ? Bm[(long long)(k0 + bnr)*ldb + gn] : 0.0f);
    }
    return h;
  }
}

template<bool BT, int ACT>
__global__ __launch_bounds__(256)
void gemm_wmma_kernel(const float* __restrict__ A, int lda, long long sAo, long long sAi,
                      const float* __restrict__ Bm, int ldb, long long sBo, long long sBi,
                      float* __restrict__ C, int ldc, long long sCo, long long sCi,
                      int M, int N, int K,
                      const float* __restrict__ bias, float alpha, int innerB)
{
  __shared__ _Float16 As [2][128][40];   // [buf][m][k]
  __shared__ _Float16 Bst[2][64][40];    // [buf][n][k]  (B staged transposed)

  const int z  = blockIdx.z;
  const int zo = z / innerB, zi = z - zo*innerB;
  A  += (long long)zo*sAo + (long long)zi*sAi;
  Bm += (long long)zo*sBo + (long long)zi*sBi;
  C  += (long long)zo*sCo + (long long)zi*sCi;

  const int m0  = blockIdx.y * 128;
  const int n0  = blockIdx.x * 64;
  const int tid = threadIdx.x;
  const int wave = tid >> 5;
  const int lane = tid & 31;
  const int wm = wave >> 1;          // 4 row-groups of 32
  const int wn = wave & 1;           // 2 col-groups of 32
  const int lm = lane & 15;
  const int ko = (lane >> 4) * 8;    // fragment k-offset within 16

  v8f c00 = {}; v8f c01 = {}; v8f c10 = {}; v8f c11 = {};

  // staging assignments (fixed per thread)
  const int ar  = tid >> 1;            // A row 0..127
  const int ak  = (tid & 1) * 16;      // A k-chunk 0 or 16
  const int btn = tid >> 2;            // BT: n row 0..63
  const int btk = (tid & 3) * 8;       // BT: k-chunk
  const int bnr = tid >> 3;            // BN: k row 0..31
  const int bnn = (tid & 7) * 8;       // BN: n-chunk

  const int bsn = BT ? btn : 0;        // (names for stores below)

  // ---- prologue: stage k-tile 0 into buffer 0
  {
    v8h a0, a1;
    fetchA(A, lda, M, m0, ar, ak, 0, K, a0, a1);
    *(v8h*)&As[0][ar][ak]     = a0;
    *(v8h*)&As[0][ar][ak + 8] = a1;
    v8h bh = fetchB<BT>(Bm, ldb, N, n0, btn, btk, bnr, bnn, 0, K);
    if (BT) {
      *(v8h*)&Bst[0][bsn][btk] = bh;
    } else {
#pragma unroll
      for (int j = 0; j < 8; ++j) Bst[0][bnn + j][bnr] = bh[j];
    }
  }
  __syncthreads();

  const int nsteps = K >> 5;           // K is a multiple of 32 for all call sites
  for (int i = 0; i < nsteps; ++i) {
    const int cur = i & 1, nxt = cur ^ 1;
    const bool hasNext = (i + 1 < nsteps);

    // ---- issue next-tile global fetch early (overlaps with WMMA below)
    v8h na0, na1, nb;
    if (hasNext) {
      fetchA(A, lda, M, m0, ar, ak, (i + 1) * 32, K, na0, na1);
      nb = fetchB<BT>(Bm, ldb, N, n0, btn, btk, bnr, bnn, (i + 1) * 32, K);
    }

    // ---- fragments: each 8-half group is one aligned ds_load_b128
    const int am0  = wm*32 + lm;
    const int bn0g = wn*32 + lm;
    V16U a0, a1, b0, b1;
    a0.h[0] = *(const v8h*)&As[cur][am0][ko];
    a0.h[1] = *(const v8h*)&As[cur][am0][16 + ko];
    a1.h[0] = *(const v8h*)&As[cur][am0 + 16][ko];
    a1.h[1] = *(const v8h*)&As[cur][am0 + 16][16 + ko];
    b0.h[0] = *(const v8h*)&Bst[cur][bn0g][ko];
    b0.h[1] = *(const v8h*)&Bst[cur][bn0g][16 + ko];
    b1.h[0] = *(const v8h*)&Bst[cur][bn0g + 16][ko];
    b1.h[1] = *(const v8h*)&Bst[cur][bn0g + 16][16 + ko];

    c00 = __builtin_amdgcn_wmma_f32_16x16x32_f16(false, a0.v, false, b0.v, (short)0, c00, false, false);
    c01 = __builtin_amdgcn_wmma_f32_16x16x32_f16(false, a0.v, false, b1.v, (short)0, c01, false, false);
    c10 = __builtin_amdgcn_wmma_f32_16x16x32_f16(false, a1.v, false, b0.v, (short)0, c10, false, false);
    c11 = __builtin_amdgcn_wmma_f32_16x16x32_f16(false, a1.v, false, b1.v, (short)0, c11, false, false);

    // ---- write next tile into the alternate buffer, then ONE barrier
    if (hasNext) {
      *(v8h*)&As[nxt][ar][ak]     = na0;
      *(v8h*)&As[nxt][ar][ak + 8] = na1;
      if (BT) {
        *(v8h*)&Bst[nxt][bsn][btk] = nb;
      } else {
#pragma unroll
        for (int j = 0; j < 8; ++j) Bst[nxt][bnn + j][bnr] = nb[j];
      }
    }
    __syncthreads();
  }

  // ---- epilogue: C layout = col (lane&15), rows v + 8*(lane>>4)
  const int col0 = n0 + wn*32 + lm;
  const int col1 = col0 + 16;
  const int rbase = m0 + wm*32 + 8*(lane >> 4);
  const float bv0 = (bias && col0 < N) ? bias[col0] : 0.0f;
  const float bv1 = (bias && col1 < N) ? bias[col1] : 0.0f;
#pragma unroll
  for (int v = 0; v < 8; ++v) {
    int r0 = rbase + v;
    int r1 = r0 + 16;
    if (r0 < M) {
      if (col0 < N) C[(long long)r0*ldc + col0] = act_apply<ACT>(c00[v]*alpha + bv0);
      if (col1 < N) C[(long long)r0*ldc + col1] = act_apply<ACT>(c01[v]*alpha + bv1);
    }
    if (r1 < M) {
      if (col0 < N) C[(long long)r1*ldc + col0] = act_apply<ACT>(c10[v]*alpha + bv0);
      if (col1 < N) C[(long long)r1*ldc + col1] = act_apply<ACT>(c11[v]*alpha + bv1);
    }
  }
}

static void launch_gemm(hipStream_t st,
    const float* A, int lda, long long sAo, long long sAi,
    const float* Bm, int ldb, long long sBo, long long sBi,
    float* C, int ldc, long long sCo, long long sCi,
    int M, int N, int K, const float* bias, float alpha,
    int outer, int inner, bool bt, int act)
{
  dim3 grid((N + 63)/64, (M + 127)/128, outer*inner);
  dim3 block(256);
  if (bt) {
    if (act == 1) gemm_wmma_kernel<true,1><<<grid,block,0,st>>>(A,lda,sAo,sAi,Bm,ldb,sBo,sBi,C,ldc,sCo,sCi,M,N,K,bias,alpha,inner);
    else          gemm_wmma_kernel<true,0><<<grid,block,0,st>>>(A,lda,sAo,sAi,Bm,ldb,sBo,sBi,C,ldc,sCo,sCi,M,N,K,bias,alpha,inner);
  } else {
    if (act == 1) gemm_wmma_kernel<false,1><<<grid,block,0,st>>>(A,lda,sAo,sAi,Bm,ldb,sBo,sBi,C,ldc,sCo,sCi,M,N,K,bias,alpha,inner);
    else          gemm_wmma_kernel<false,0><<<grid,block,0,st>>>(A,lda,sAo,sAi,Bm,ldb,sBo,sBi,C,ldc,sCo,sCi,M,N,K,bias,alpha,inner);
  }
}

// ======================= GAT kernels =======================
__device__ inline void atomicMaxFloat(float* addr, float val){
  unsigned int* ua = (unsigned int*)addr;
  unsigned int old = *ua;
  while (true) {
    float f = __uint_as_float(old);
    if (f >= val) break;
    unsigned int assumed = old;
    old = atomicCAS(ua, assumed, __float_as_uint(val));
    if (old == assumed) break;
  }
}
__device__ inline float leaky02(float x){ return x > 0.0f ? x : 0.2f*x; }

__global__ void gat_coef_kernel(const float* __restrict__ Hf,
                                const float* __restrict__ a_src,
                                const float* __restrict__ a_dst,
                                float* asrc, float* adst, int Nn, int Hh, int C){
  int i = blockIdx.x*blockDim.x + threadIdx.x;
  if (i >= Nn*Hh) return;
  int n = i / Hh, h = i - n*Hh;
  const float* hp = Hf + (long long)n*Hh*C + h*C;
  float s1 = 0.f, s2 = 0.f;
  for (int c = 0; c < C; ++c) { float hv = hp[c]; s1 += hv*a_src[h*C+c]; s2 += hv*a_dst[h*C+c]; }
  asrc[i] = s1; adst[i] = s2;
}
__global__ void edge_max_kernel(const int* __restrict__ src, const int* __restrict__ dst,
                                int E, int Nn, const float* __restrict__ asrc,
                                const float* __restrict__ adst, float* mmax, int Hh){
  long long i = (long long)blockIdx.x*blockDim.x + threadIdx.x;
  long long tot = (long long)(E + Nn)*Hh;
  if (i >= tot) return;
  long long e = i / Hh; int h = (int)(i - e*Hh);
  int s = (e < E) ? src[e] : (int)(e - E);
  int d = (e < E) ? dst[e] : (int)(e - E);
  float ev = leaky02(asrc[s*Hh + h] + adst[d*Hh + h]);
  atomicMaxFloat(&mmax[d*Hh + h], ev);
}
__global__ void edge_sum_kernel(const int* __restrict__ src, const int* __restrict__ dst,
                                int E, int Nn, const float* __restrict__ asrc,
                                const float* __restrict__ adst, const float* __restrict__ mmax,
                                float* ssum, int Hh){
  long long i = (long long)blockIdx.x*blockDim.x + threadIdx.x;
  long long tot = (long long)(E + Nn)*Hh;
  if (i >= tot) return;
  long long e = i / Hh; int h = (int)(i - e*Hh);
  int s = (e < E) ? src[e] : (int)(e - E);
  int d = (e < E) ? dst[e] : (int)(e - E);
  float ev = leaky02(asrc[s*Hh + h] + adst[d*Hh + h]);
  atomicAdd(&ssum[d*Hh + h], __expf(ev - mmax[d*Hh + h]));
}
__global__ void edge_agg_kernel(const int* __restrict__ src, const int* __restrict__ dst,
                                int E, int Nn, const float* __restrict__ asrc,
                                const float* __restrict__ adst, const float* __restrict__ Hf,
                                const float* __restrict__ mmax, const float* __restrict__ ssum,
                                float* out, int Hh, int C){
  long long i = (long long)blockIdx.x*blockDim.x + threadIdx.x;
  long long tot = (long long)(E + Nn)*Hh;
  if (i >= tot) return;
  long long e = i / Hh; int h = (int)(i - e*Hh);
  int s = (e < E) ? src[e] : (int)(e - E);
  int d = (e < E) ? dst[e] : (int)(e - E);
  float ev = leaky02(asrc[s*Hh + h] + adst[d*Hh + h]);
  float alpha = __expf(ev - mmax[d*Hh + h]) / ssum[d*Hh + h];
  const float* hp = Hf + (long long)s*Hh*C + h*C;
  float* op = out + (long long)d*Hh*C + h*C;
  for (int c = 0; c < C; ++c) atomicAdd(&op[c], hp[c]*alpha);
}

// ======================= pack / transformer support =======================
__global__ void count_kernel(const int* __restrict__ batch, int* counts, int Nn){
  int i = blockIdx.x*blockDim.x + threadIdx.x;
  if (i < Nn) atomicAdd(&counts[batch[i]], 1);
}
__global__ void scan_offsets_kernel(const int* __restrict__ counts, int* offsets){
  if (threadIdx.x == 0) { int acc = 0; for (int b = 0; b < Bc; ++b) { offsets[b] = acc; acc += counts[b]; } }
}
__global__ void pack_scatter_kernel(const float* __restrict__ x, const int* __restrict__ batch,
                                    const int* __restrict__ offsets, float* padded, int Nn){
  long long i = (long long)blockIdx.x*blockDim.x + threadIdx.x;
  if (i >= (long long)Nn*Dc) return;
  int n = (int)(i / Dc), d = (int)(i - (long long)n*Dc);
  int b = batch[n];
  int pos = n - offsets[b];
  if (pos < MAXNc) padded[((long long)b*MAXNc + pos)*Dc + d] = x[i];
}
__global__ void make_valid_kernel(const int* __restrict__ counts, int* valid){
  int i = blockIdx.x*blockDim.x + threadIdx.x;
  if (i >= Bc*MAXNc) return;
  int b = i / MAXNc, j = i - b*MAXNc;
  int c = counts[b]; if (c > MAXNc) c = MAXNc;
  valid[i] = (j < c) ? 1 : 0;
}
// y = LayerNorm(a + r) * g + be, rows of Dc; one block (256 thr) per row
__global__ void ln_res_kernel(const float* __restrict__ a, const float* __restrict__ r,
                              const float* __restrict__ g, const float* __restrict__ be,
                              float* out){
  int row = blockIdx.x, t = threadIdx.x;
  float v = a[(long long)row*Dc + t] + r[(long long)row*Dc + t];
  __shared__ float red[256];
  red[t] = v; __syncthreads();
  for (int s = 128; s > 0; s >>= 1) { if (t < s) red[t] += red[t+s]; __syncthreads(); }
  float mean = red[0] * (1.0f/Dc); __syncthreads();
  float dv = v - mean;
  red[t] = dv*dv; __syncthreads();
  for (int s = 128; s > 0; s >>= 1) { if (t < s) red[t] += red[t+s]; __syncthreads(); }
  float var = red[0] * (1.0f/Dc);
  out[(long long)row*Dc + t] = dv * rsqrtf(var + 1e-5f) * g[t] + be[t];
}
// masked softmax over rows of length MAXNc(=512); grid = B*H*Lq, block 256
__global__ void masked_softmax_kernel(float* scores, const int* __restrict__ valid){
  long long row = blockIdx.x;
  int b = (int)(row / ((long long)Hc*MAXNc));
  float* sp = scores + row*MAXNc;
  const int* vp = valid + (long long)b*MAXNc;
  int t = threadIdx.x;
  float x0 = vp[t]       ? sp[t]       : -1e30f;
  float x1 = vp[t + 256] ? sp[t + 256] : -1e30f;
  __shared__ float red[256];
  red[t] = fmaxf(x0, x1); __syncthreads();
  for (int s = 128; s > 0; s >>= 1) { if (t < s) red[t] = fmaxf(red[t], red[t+s]); __syncthreads(); }
  float mx = red[0]; __syncthreads();
  float e0 = __expf(x0 - mx), e1 = __expf(x1 - mx);
  red[t] = e0 + e1; __syncthreads();
  for (int s = 128; s > 0; s >>= 1) { if (t < s) red[t] += red[t+s]; __syncthreads(); }
  float inv = 1.0f / red[0];
  sp[t] = e0*inv; sp[t + 256] = e1*inv;
}
__global__ void gather_h0_kernel(const float* __restrict__ tgt, float* h0){
  int i = blockIdx.x*blockDim.x + threadIdx.x;
  if (i >= Bc*Dc) return;
  int b = i / Dc, d = i - b*Dc;
  h0[i] = tgt[(long long)b*MAXNc*Dc + d];
}
__global__ void final_out_kernel(const float* __restrict__ h, const float* __restrict__ w,
                                 const float* __restrict__ bia, float* out){
  int i = blockIdx.x*blockDim.x + threadIdx.x;
  if (i >= Bc*2) return;
  int b = i >> 1, o = i & 1;
  float s = bia[o];
  for (int d = 0; d < Dc; ++d) s += h[b*Dc + d] * w[o*Dc + d];
  out[i] = s;
}

// ======================= host orchestration =======================
static inline int cdiv(long long a, int b){ return (int)((a + b - 1)/b); }

static void gat_layer(hipStream_t st, const float* xin, int inF, int Hh, int C,
                      const float* W, const float* a_src, const float* a_dst, const float* bias,
                      const int* esrc, const int* edst, int E, int Nn, int act,
                      float* hbuf, float* asrc, float* adst, float* mbuf, float* sbuf, float* obuf)
{
  const int F = Hh*C;
  // h = x @ W   (W stored KxN row-major -> BT=false)
  launch_gemm(st, xin, inF, 0,0, W, F, 0,0, hbuf, F, 0,0, Nn, F, inF, nullptr, 1.0f, 1,1, false, 0);
  int tot = Nn*Hh;
  gat_coef_kernel<<<cdiv(tot,256),256,0,st>>>(hbuf, a_src, a_dst, asrc, adst, Nn, Hh, C);
  fill_f_kernel<<<cdiv((long long)Nn*Hh,256),256,0,st>>>(mbuf, -1e30f, (long long)Nn*Hh);
  fill_f_kernel<<<cdiv((long long)Nn*Hh,256),256,0,st>>>(sbuf, 0.0f, (long long)Nn*Hh);
  fill_f_kernel<<<cdiv((long long)Nn*F,256),256,0,st>>>(obuf, 0.0f, (long long)Nn*F);
  long long et = (long long)(E + Nn)*Hh;
  edge_max_kernel<<<cdiv(et,256),256,0,st>>>(esrc, edst, E, Nn, asrc, adst, mbuf, Hh);
  edge_sum_kernel<<<cdiv(et,256),256,0,st>>>(esrc, edst, E, Nn, asrc, adst, mbuf, sbuf, Hh);
  edge_agg_kernel<<<cdiv(et,256),256,0,st>>>(esrc, edst, E, Nn, asrc, adst, hbuf, mbuf, sbuf, obuf, Hh, C);
  bias_act_kernel<<<cdiv((long long)Nn*F,256),256,0,st>>>(obuf, bias, (long long)Nn*F, F, act);
}

struct WSBufs {
  float *qkv, *qbuf, *kvbuf, *scores, *obuf, *mho, *ffnh;
};

static void mha(hipStream_t st, const float* qin, const float* kvin,
                const float* in_w, const float* in_b,
                const float* out_w, const float* out_b,
                const int* valid_keys, float* outbuf, const WSBufs& w, bool self)
{
  const float *qp, *kp, *vp; int ldq, ldkv;
  if (self) {
    launch_gemm(st, qin, Dc, 0,0, in_w, Dc, 0,0, w.qkv, 3*Dc, 0,0, TQc, 3*Dc, Dc, in_b, 1.0f, 1,1, true, 0);
    qp = w.qkv; ldq = 3*Dc; kp = w.qkv + Dc; vp = w.qkv + 2*Dc; ldkv = 3*Dc;
  } else {
    launch_gemm(st, qin,  Dc, 0,0, in_w,        Dc, 0,0, w.qbuf,  Dc,  0,0, TQc, Dc,   Dc, in_b,      1.0f, 1,1, true, 0);
    launch_gemm(st, kvin, Dc, 0,0, in_w + Dc*Dc,Dc, 0,0, w.kvbuf, 2*Dc,0,0, TQc, 2*Dc, Dc, in_b + Dc, 1.0f, 1,1, true, 0);
    qp = w.qbuf; ldq = Dc; kp = w.kvbuf; vp = w.kvbuf + Dc; ldkv = 2*Dc;
  }
  // scores[b,h] = (1/sqrt(DH)) * Q[b,:,h,:] @ K[b,:,h,:]^T
  launch_gemm(st, qp, ldq, (long long)MAXNc*ldq, DHc,
                  kp, ldkv,(long long)MAXNc*ldkv, DHc,
                  w.scores, MAXNc, (long long)Hc*MAXNc*MAXNc, (long long)MAXNc*MAXNc,
                  MAXNc, MAXNc, DHc, nullptr, 0.1767766953f, Bc, Hc, true, 0);
  masked_softmax_kernel<<<Bc*Hc*MAXNc, 256, 0, st>>>(w.scores, valid_keys);
  // o[b,:,h,:] = attn @ V
  launch_gemm(st, w.scores, MAXNc, (long long)Hc*MAXNc*MAXNc, (long long)MAXNc*MAXNc,
                  vp, ldkv, (long long)MAXNc*ldkv, DHc,
                  w.obuf, Dc, (long long)MAXNc*Dc, DHc,
                  MAXNc, DHc, MAXNc, nullptr, 1.0f, Bc, Hc, false, 0);
  launch_gemm(st, w.obuf, Dc, 0,0, out_w, Dc, 0,0, outbuf, Dc, 0,0, TQc, Dc, Dc, out_b, 1.0f, 1,1, true, 0);
}

static void ffn(hipStream_t st, const float* xin, const float* w1, const float* b1,
                const float* w2, const float* b2, float* outbuf, const WSBufs& w)
{
  launch_gemm(st, xin,   Dc,  0,0, w1, Dc,  0,0, w.ffnh, FFc, 0,0, TQc, FFc, Dc,  b1, 1.0f, 1,1, true, 1);
  launch_gemm(st, w.ffnh,FFc, 0,0, w2, FFc, 0,0, outbuf, Dc,  0,0, TQc, Dc,  FFc, b2, 1.0f, 1,1, true, 0);
}

extern "C" void kernel_launch(void* const* d_in, const int* in_sizes, int n_in,
                              void* d_out, int out_size, void* d_ws, size_t ws_size,
                              hipStream_t stream)
{
  #define PF(i) ((const float*)d_in[(i)])
  const float* x    = PF(0);
  const int* ei     = (const int*)d_in[1];
  const int* batch  = (const int*)d_in[2];
  const float* x_t  = PF(3);
  const int* ei_t   = (const int*)d_in[4];
  const int* batch_t= (const int*)d_in[5];
  const int Nn = in_sizes[0] / 32;
  const int E  = in_sizes[1] / 2;

  // ---- workspace bump allocator ----
  char* wp = (char*)d_ws;
  auto alloc_f = [&](long long elems) -> float* {
    float* p = (float*)wp; wp += ((elems*4 + 255) & ~255LL); return p; };
  auto alloc_i = [&](long long elems) -> int* {
    int* p = (int*)wp; wp += ((elems*4 + 255) & ~255LL); return p; };

  float* xg_c = alloc_f((long long)Nn*Dc);
  float* xg_t = alloc_f((long long)Nn*Dc);
  float* h1   = alloc_f((long long)Nn*320);
  float* as_  = alloc_f((long long)Nn*10);
  float* ad_  = alloc_f((long long)Nn*10);
  float* m_   = alloc_f((long long)Nn*10);
  float* s_   = alloc_f((long long)Nn*10);
  float* o1   = alloc_f((long long)Nn*320);
  float* h2   = alloc_f((long long)Nn*128);
  float* o2   = alloc_f((long long)Nn*128);
  int* counts  = alloc_i(Bc);
  int* offsets = alloc_i(Bc);
  float* compound = alloc_f((long long)TQc*Dc);
  float* target   = alloc_f((long long)TQc*Dc);
  int* valid_c = alloc_i(TQc);
  int* valid_t = alloc_i(TQc);
  WSBufs w;
  w.qkv    = alloc_f((long long)TQc*3*Dc);
  w.qbuf   = alloc_f((long long)TQc*Dc);
  w.kvbuf  = alloc_f((long long)TQc*2*Dc);
  w.scores = alloc_f((long long)Bc*Hc*MAXNc*MAXNc);
  w.obuf   = alloc_f((long long)TQc*Dc);
  w.mho    = alloc_f((long long)TQc*Dc);
  w.ffnh   = alloc_f((long long)TQc*FFc);
  float* h0   = alloc_f(Bc*Dc);
  float* fc1o = alloc_f(Bc*Dc);

  // ---- graph branches (params shared between branches) ----
  auto graph_branch = [&](const float* xin, const int* eidx, float* xg){
    const int* esrc = eidx; const int* edst = eidx + E;
    gat_layer(stream, xin, 32, 10, 32, PF(6), PF(7), PF(8), PF(9),
              esrc, edst, E, Nn, /*ELU*/2, h1, as_, ad_, m_, s_, o1);
    gat_layer(stream, o1, 320, 1, 128, PF(10), PF(11), PF(12), PF(13),
              esrc, edst, E, Nn, /*ReLU*/1, h2, as_, ad_, m_, s_, o2);
    // fcg1: relu(o2 @ fcg1_w.T + b), fcg1_w is (256,128) -> BT
    launch_gemm(stream, o2, 128, 0,0, PF(14), 128, 0,0, xg, Dc, 0,0,
                Nn, Dc, 128, PF(15), 1.0f, 1,1, true, 1);
  };
  graph_branch(x,   ei,   xg_c);
  graph_branch(x_t, ei_t, xg_t);

  // ---- pack ----
  auto pack = [&](const float* xg, const int* bvec, float* padded, int* valid){
    fill_i_kernel<<<1, 32, 0, stream>>>(counts, 0, Bc);
    count_kernel<<<cdiv(Nn,256),256,0,stream>>>(bvec, counts, Nn);
    scan_offsets_kernel<<<1, 32, 0, stream>>>(counts, offsets);
    fill_f_kernel<<<cdiv((long long)TQc*Dc,256),256,0,stream>>>(padded, 0.0f, (long long)TQc*Dc);
    pack_scatter_kernel<<<cdiv((long long)Nn*Dc,256),256,0,stream>>>(xg, bvec, offsets, padded, Nn);
    make_valid_kernel<<<cdiv(TQc,256),256,0,stream>>>(counts, valid);
  };
  pack(xg_c, batch,   compound, valid_c);
  pack(xg_t, batch_t, target,   valid_t);

  // ---- encoder (src = target) ----
  float* src = target;
  const int encBase[2] = {16, 46};
  for (int i = 0; i < 2; ++i) {
    int p = encBase[i];
    mha(stream, src, src, PF(p+0), PF(p+1), PF(p+2), PF(p+3), valid_t, w.mho, w, true);
    ln_res_kernel<<<TQc,256,0,stream>>>(w.mho, src, PF(p+8), PF(p+9), src);
    ffn(stream, src, PF(p+4), PF(p+5), PF(p+6), PF(p+7), w.mho, w);
    ln_res_kernel<<<TQc,256,0,stream>>>(w.mho, src, PF(p+10), PF(p+11), src);
  }

  // ---- decoder (tgt = compound) ----
  float* tgt = compound;
  const int decBase[2] = {28, 58};
  for (int i = 0; i < 2; ++i) {
    int p = decBase[i];
    // self attn (keys = compound -> valid_c)
    mha(stream, tgt, tgt, PF(p+0), PF(p+1), PF(p+2), PF(p+3), valid_c, w.mho, w, true);
    ln_res_kernel<<<TQc,256,0,stream>>>(w.mho, tgt, PF(p+12), PF(p+13), tgt);
    // cross attn (keys = src -> valid_t)
    mha(stream, tgt, src, PF(p+4), PF(p+5), PF(p+6), PF(p+7), valid_t, w.mho, w, false);
    ln_res_kernel<<<TQc,256,0,stream>>>(w.mho, tgt, PF(p+14), PF(p+15), tgt);
    ffn(stream, tgt, PF(p+8), PF(p+9), PF(p+10), PF(p+11), w.mho, w);
    ln_res_kernel<<<TQc,256,0,stream>>>(w.mho, tgt, PF(p+16), PF(p+17), tgt);
  }

  // ---- head ----
  gather_h0_kernel<<<cdiv(Bc*Dc,256),256,0,stream>>>(tgt, h0);
  launch_gemm(stream, h0, Dc, 0,0, PF(76), Dc, 0,0, fc1o, Dc, 0,0,
              Bc, Dc, Dc, PF(77), 1.0f, 1,1, true, 1);
  final_out_kernel<<<1, 64, 0, stream>>>(fc1o, PF(78), PF(79), (float*)d_out);
  #undef PF
}
